// MemoryEfficientAttention_50079318671753
// MI455X (gfx1250) — matrix operand order repaired and checked by reading, not compiled, not run on hardware
//
#include <hip/hip_runtime.h>

// Flash-attention-2 causal attention for MI455X (gfx1250, wave32).
// B=2, H=16, S=2048, D=64, fp32 in/out; matmuls on f16 WMMA with f32 accum.
// Row-sum of P computed by WMMA against an all-ones B matrix (no shuffles);
// only the row-max needs a 4-stage lane butterfly.

typedef __attribute__((ext_vector_type(16))) _Float16 v16h;
typedef __attribute__((ext_vector_type(8)))  _Float16 v8h;
typedef __attribute__((ext_vector_type(2)))  __fp16   v2y;   // cvt_pkrtz result type
typedef __attribute__((ext_vector_type(4)))  __fp16   v4y;
typedef __attribute__((ext_vector_type(8)))  float    v8f;
typedef __attribute__((ext_vector_type(4)))  float    v4f;

#define S_LEN  2048
#define D_DIM  64
#define QTILE  64   // q rows per workgroup (4 waves x 16)
#define KVTILE 64   // keys per inner iteration

static __device__ __forceinline__ v8f wmma_f16(v16h a, v16h b, v8f c) {
  // D = A(16x32 f16) * B(32x16 f16) + C(16x16 f32)
  return __builtin_amdgcn_wmma_f32_16x16x32_f16(false, a, false, b,
                                                (short)0, c, false, false);
}

static __device__ __forceinline__ float fast_exp2(float x) {
#if __has_builtin(__builtin_amdgcn_exp2f)
  return __builtin_amdgcn_exp2f(x);
#else
  return exp2f(x);
#endif
}

static __device__ __forceinline__ v16h cat8(v8h lo, v8h hi) {
  return __builtin_shufflevector(lo, hi, 0,1,2,3,4,5,6,7,8,9,10,11,12,13,14,15);
}

__global__ __launch_bounds__(256, 1)
void fa2_causal_wmma(const float* __restrict__ Q, const float* __restrict__ K,
                     const float* __restrict__ V, float* __restrict__ Out) {
  // LDS: K tile row-major [s][d], V tile transposed [d][s], per-wave P scratch.
  __shared__ __align__(32) _Float16 kT[KVTILE * D_DIM];       // 8 KB
  __shared__ __align__(32) _Float16 vT[D_DIM * KVTILE];       // 8 KB
  __shared__ __align__(32) _Float16 pT[4][16 * KVTILE];       // 8 KB

  const int tid   = threadIdx.x;
  const int wave  = tid >> 5;
  const int lane  = tid & 31;
  const int lhalf = lane >> 4;   // 0: lanes 0-15, 1: lanes 16-31
  const int l16   = lane & 15;

  const int nqt = S_LEN / QTILE;
  const int bh  = blockIdx.x / nqt;
  const int q0  = (blockIdx.x % nqt) * QTILE;

  const size_t base = (size_t)bh * S_LEN * D_DIM;
  const float* Qp = Q + base;
  const float* Kp = K + base;
  const float* Vp = V + base;
  float*       Op = Out + base;

  // ---- Load Q A-fragments; fold (1/sqrt(64))*log2(e) so softmax runs in exp2 ----
  // A 16x32 fragment in this code's contiguous k order (B uses the same): lane<16 -> row l16, K = [0,16); lane>=16 -> [16,32)
  const float qscale = 0.125f * 1.44269504f;
  const int   qrowg  = q0 + wave * 16 + l16;
  v16h qa[2];
  {
    const int klo = lhalf ? 16 : 0;
    for (int f = 0; f < 2; ++f) {
      const float* qr = Qp + (size_t)qrowg * D_DIM + f * 32;
      v4f x0 = *(const v4f*)(qr + klo);
      v4f x1 = *(const v4f*)(qr + klo + 4);
      v4f x2 = *(const v4f*)(qr + klo + 8);
      v4f x3 = *(const v4f*)(qr + klo + 12);
      for (int i = 0; i < 4; ++i) {
        qa[f][i]      = (_Float16)(x0[i] * qscale);
        qa[f][4 + i]  = (_Float16)(x1[i] * qscale);
        qa[f][8 + i]  = (_Float16)(x2[i] * qscale);
        qa[f][12 + i] = (_Float16)(x3[i] * qscale);
      }
    }
  }

  v16h ones;
  for (int i = 0; i < 16; ++i) ones[i] = (_Float16)1.0f;

  // ---- Online-softmax state (C-layout: VGPR r holds row lhalf*8 + r) ----
  float mstate[8];
  v8f   oacc[4], lfrag;
  for (int r = 0; r < 8; ++r) mstate[r] = -3.0e38f;
  for (int dt = 0; dt < 4; ++dt) oacc[dt] = {};
  lfrag = {};

  const int rowbase  = q0 + wave * 16 + lhalf * 8;
  const int kv_iters = (q0 + QTILE) / KVTILE;   // = q-tile index + 1

  for (int it = 0; it < kv_iters; ++it) {
    const int kv0 = it * KVTILE;

    __syncthreads();  // previous iteration's LDS reads done
    // ---- Cooperative stage: K/V[kv0..+64][0..64) fp32 -> f16 LDS ----
    for (int j = 0; j < 4; ++j) {
      const int i  = tid + j * 256;        // 1024 float4 chunks
      const int s  = i >> 4;               // 0..63 key within tile
      const int d4 = (i & 15) * 4;         // 0..60
      v4f kx = *(const v4f*)(Kp + (size_t)(kv0 + s) * D_DIM + d4);
      v4f vx = *(const v4f*)(Vp + (size_t)(kv0 + s) * D_DIM + d4);
      v2y klo2 = __builtin_amdgcn_cvt_pkrtz(kx[0], kx[1]);
      v2y khi2 = __builtin_amdgcn_cvt_pkrtz(kx[2], kx[3]);
      *(v4y*)&kT[s * D_DIM + d4] = __builtin_shufflevector(klo2, khi2, 0, 1, 2, 3);
      vT[(d4 + 0) * KVTILE + s] = (_Float16)vx[0];
      vT[(d4 + 1) * KVTILE + s] = (_Float16)vx[1];
      vT[(d4 + 2) * KVTILE + s] = (_Float16)vx[2];
      vT[(d4 + 3) * KVTILE + s] = (_Float16)vx[3];
    }
    __syncthreads();

    // Prefetch next KV tile (global_prefetch_b8)
    if (it + 1 < kv_iters) {
      const size_t noff = (size_t)(kv0 + KVTILE + (tid >> 2)) * D_DIM + (tid & 3) * 16;
      __builtin_prefetch(Kp + noff, 0, 1);
      __builtin_prefetch(Vp + noff, 0, 1);
    }

    // ---- S = Q @ K^T : 4 N-tiles x (K-dim 64 in 2 steps) = 8 WMMAs ----
    v8f sfrag[4];
    for (int nt = 0; nt < 4; ++nt) {
      const int ks = nt * 16 + l16;               // key = B column n
      v8f c = {};
      for (int f = 0; f < 2; ++f) {
        const v16h* bp = (const v16h*)&kT[ks * D_DIM + f * 32 + lhalf * 16];
        c = wmma_f16(qa[f], *bp, c);
      }
      sfrag[nt] = c;
    }

    // ---- Causal mask (only tiles straddling/above the diagonal) ----
    for (int nt = 0; nt < 4; ++nt) {
      if (kv0 + nt * 16 + 15 > q0 + wave * 16) {
        const int gs = kv0 + nt * 16 + l16;
        for (int r = 0; r < 8; ++r)
          if (gs > rowbase + r) sfrag[nt][r] = -3.0e38f;
      }
    }

    // ---- Row max (butterfly over 16 lanes) + exp2; row-sum comes via WMMA ----
    float alpha[8];
    for (int r = 0; r < 8; ++r) {
      float mx = fmaxf(fmaxf(sfrag[0][r], sfrag[1][r]),
                       fmaxf(sfrag[2][r], sfrag[3][r]));
      mx = fmaxf(mx, __shfl_xor(mx, 1, 32));
      mx = fmaxf(mx, __shfl_xor(mx, 2, 32));
      mx = fmaxf(mx, __shfl_xor(mx, 4, 32));
      mx = fmaxf(mx, __shfl_xor(mx, 8, 32));
      const float mnew = fmaxf(mstate[r], mx);
      alpha[r]  = fast_exp2(mstate[r] - mnew);
      mstate[r] = mnew;
      for (int nt = 0; nt < 4; ++nt)
        sfrag[nt][r] = fast_exp2(sfrag[nt][r] - mnew);
    }

    // ---- P: C-layout -> LDS (row-major 16x64); rescale accumulators ----
    _Float16* pw = &pT[wave][0];
    for (int r = 0; r < 8; ++r) {
      const int m = lhalf * 8 + r;
      pw[m * KVTILE +      l16] = (_Float16)sfrag[0][r];
      pw[m * KVTILE + 16 + l16] = (_Float16)sfrag[1][r];
      pw[m * KVTILE + 32 + l16] = (_Float16)sfrag[2][r];
      pw[m * KVTILE + 48 + l16] = (_Float16)sfrag[3][r];
      lfrag[r] *= alpha[r];
      for (int dt = 0; dt < 4; ++dt) oacc[dt][r] *= alpha[r];
    }

    // ---- Read P as two A-fragments (K = 0..31, 32..63) ----
    v16h pa[2];
    {
      const int klo = lhalf ? 16 : 0;
      for (int f = 0; f < 2; ++f) {
        const v8h* a0 = (const v8h*)&pw[l16 * KVTILE + f * 32 + klo];
        const v8h* a1 = (const v8h*)&pw[l16 * KVTILE + f * 32 + klo + 8];
        pa[f] = cat8(*a0, *a1);
      }
    }

    // ---- l += P @ ones : row-sum on the matrix pipe (2 WMMAs) ----
    lfrag = wmma_f16(pa[0], ones, lfrag);
    lfrag = wmma_f16(pa[1], ones, lfrag);

    // ---- O += P @ V : 4 d-tiles x (K-dim 64 in 2 steps) = 8 WMMAs ----
    for (int dt = 0; dt < 4; ++dt) {
      const int d = dt * 16 + l16;                // B column n = d
      v8f c = oacc[dt];
      for (int f = 0; f < 2; ++f) {
        const v16h* bp = (const v16h*)&vT[d * KVTILE + f * 32 + lhalf * 16];
        c = wmma_f16(pa[f], *bp, c);
      }
      oacc[dt] = c;
    }
  }

  // ---- Normalize and write out (coalesced per half-wave) ----
  for (int r = 0; r < 8; ++r) {
    const int gm = rowbase + r;
    const float inv = 1.0f / lfrag[r];
    for (int dt = 0; dt < 4; ++dt)
      Op[(size_t)gm * D_DIM + dt * 16 + l16] = oacc[dt][r] * inv;
  }
}

extern "C" void kernel_launch(void* const* d_in, const int* in_sizes, int n_in,
                              void* d_out, int out_size, void* d_ws, size_t ws_size,
                              hipStream_t stream) {
  const float* q = (const float*)d_in[0];
  const float* k = (const float*)d_in[1];
  const float* v = (const float*)d_in[2];
  float* out = (float*)d_out;

  const int BH  = in_sizes[0] / (S_LEN * D_DIM);   // B*H = 32
  const int nqt = S_LEN / QTILE;                   // 32 q-tiles
  dim3 grid(BH * nqt);
  dim3 block(256);
  fa2_causal_wmma<<<grid, block, 0, stream>>>(q, k, v, out);
}
